// DATrans_76201309766482
// MI455X (gfx1250) — matrix operand order repaired and checked by reading, not compiled, not run on hardware
//
#include <hip/hip_runtime.h>
#include <cstdint>
#include <cstddef>

// ---------------------------------------------------------------------------
// Types for CDNA5 WMMA (wave32): v_wmma_f32_16x16x32_bf16
// ---------------------------------------------------------------------------
typedef __attribute__((ext_vector_type(16))) __bf16 bf16x16;
typedef __attribute__((ext_vector_type(8)))  float  floatx8;
typedef __attribute__((ext_vector_type(4)))  int    i32x4;

#define AREA   65536
#define HH     256
#define WW     256
#define BATCH  4
#define CIN    16

#if __has_builtin(__builtin_amdgcn_global_load_async_to_lds_b128)
#define HAVE_ASYNC_LDS 1
#else
#define HAVE_ASYNC_LDS 0
#endif

__device__ __forceinline__ int refl(int t, int n) {
  t = (t < 0) ? -t : t;
  return (t >= n) ? (2 * n - 2 - t) : t;
}

__device__ __forceinline__ floatx8 wmma_bf16(bf16x16 a, bf16x16 b, floatx8 c) {
  return __builtin_amdgcn_wmma_f32_16x16x32_bf16(false, a, false, b, (short)0, c,
                                                 false, false);
}

union FragU { bf16x16 v; uint4 q[2]; };

// A-fragment: 8 halves at base (k=kbase..kbase+7), 8 halves at base+16 (k=+16..+23)
__device__ __forceinline__ bf16x16 ld_afrag(const __bf16* p) {
  FragU u;
  u.q[0] = *(const uint4*)p;
  u.q[1] = *(const uint4*)(p + 16);
  return u.v;
}
// Contiguous 16-half (32B) fragment (B-fragments and pre-packed lane fragments)
__device__ __forceinline__ bf16x16 ld_bfrag(const __bf16* p) {
  FragU u;
  u.q[0] = *(const uint4*)p;
  u.q[1] = *(const uint4*)(p + 8);
  return u.v;
}

__device__ __forceinline__ floatx8 zero8() {
  floatx8 z = {0.f, 0.f, 0.f, 0.f, 0.f, 0.f, 0.f, 0.f};
  return z;
}

__device__ __forceinline__ float sumsq8(floatx8 v) {
  float s = 0.f;
#pragma unroll
  for (int r = 0; r < 8; ++r) s += v[r] * v[r];
  return s;
}

// Async global->LDS 16B copy (CDNA5 GLOBAL_LOAD_ASYNC_TO_LDS_B128); falls
// back to a plain copy if the builtin is not declared by this toolchain.
__device__ __forceinline__ void async_cp16(__bf16* lds_dst, const __bf16* glb_src) {
#if HAVE_ASYNC_LDS
  __builtin_amdgcn_global_load_async_to_lds_b128(
      (__attribute__((address_space(1))) i32x4*)glb_src,
      (__attribute__((address_space(3))) i32x4*)lds_dst, 0, 0);
#else
  *(uint4*)lds_dst = *(const uint4*)glb_src;
#endif
}

__device__ __forceinline__ void stage_fence() {
#if HAVE_ASYNC_LDS
#if __has_builtin(__builtin_amdgcn_s_wait_asynccnt)
  __builtin_amdgcn_s_wait_asynccnt(0);
#else
  asm volatile("s_wait_asynccnt 0x0" ::: "memory");
#endif
#endif
}

// Shift direction (branch-free arithmetic, no const-array lookups)
__device__ __forceinline__ int shift_dy(int s) {
  return (s < 3) ? -1 : ((s >= 5) ? 1 : 0);
}
__device__ __forceinline__ int shift_dx(int s) {
  if (s < 3) return s - 1;
  if (s == 3) return -1;
  if (s == 4) return 1;
  return s - 6;
}

// Build sur tile in LDS: surW[px*128 + s*16 + c] = shifted(cen) - cen, bf16.
__device__ __forceinline__ void build_sur(__bf16* surW, const __bf16* cenbf,
                                          size_t cbase, int y, int x, int px,
                                          int hi, int d, const float* cf) {
#pragma unroll
  for (int s8 = 0; s8 < 4; ++s8) {
    int s  = hi * 4 + s8;
    int ry = refl(y + shift_dy(s) * d, HH);
    int rx = refl(x + shift_dx(s) * d, WW);
    const __bf16* p = cenbf + cbase + (size_t)ry * WW + rx;
#pragma unroll
    for (int c = 0; c < 16; ++c) {
      surW[px * 128 + s * 16 + c] = (__bf16)((float)p[(size_t)c * AREA] - cf[c]);
    }
  }
}

// ---------------------------------------------------------------------------
// Kernel 1: setup — bf16 cen, fragment-packed weights, BN fold, zero accums
// ---------------------------------------------------------------------------
__global__ __launch_bounds__(256) void k_setup(
    const float* __restrict__ cen, const float* __restrict__ qw,
    const float* __restrict__ kw, const float* __restrict__ vw,
    const float* __restrict__ ow, const float* __restrict__ g,
    const float* __restrict__ bt, const float* __restrict__ mn,
    const float* __restrict__ vr, __bf16* __restrict__ cenbf,
    __bf16* __restrict__ kpack, __bf16* __restrict__ vpack,
    __bf16* __restrict__ qpack, __bf16* __restrict__ owpack,
    float* __restrict__ bns, float* __restrict__ bnb,
    float* __restrict__ qnorm, float* __restrict__ knorm,
    float* __restrict__ sraw) {
  const int tid = blockIdx.x * blockDim.x + threadIdx.x;
  const int nth = gridDim.x * blockDim.x;
  for (int i = tid; i < BATCH * CIN * AREA; i += nth) cenbf[i] = (__bf16)cen[i];
  // K/V projection weights as B-fragments: B[k][o] = w[o][k]
  for (int i = tid; i < 65536; i += nth) {
    int e = i & 15, L = (i >> 4) & 31, nt = (i >> 9) & 7, ks = (i >> 12) & 3,
        h = (i >> 14) & 3;
    int k = ks * 32 + ((L >> 4) & 1) * 8 + (e & 7) + (e >> 3) * 16;
    int o = nt * 16 + (L & 15);
    kpack[i] = (__bf16)kw[(h * 128 + o) * 128 + k];
    vpack[i] = (__bf16)vw[(h * 128 + o) * 128 + k];
  }
  // Q weights as zero-padded (K=32) B-fragments
  for (int i = tid; i < 2048; i += nth) {
    int e = i & 15, L = (i >> 4) & 31, h = (i >> 9) & 3;
    qpack[i] = (L < 16) ? (__bf16)qw[(h * 16 + L) * 16 + e] : (__bf16)0.f;
  }
  // out_w as A-fragments (M=o, K=t), 2 k-steps
  for (int i = tid; i < 1024; i += nth) {
    int e = i & 15, L = (i >> 4) & 31, ks = (i >> 9) & 1;
    int m = L & 15;
    int k = ks * 32 + ((L >> 4) & 1) * 8 + (e & 7) + (e >> 3) * 16;
    owpack[i] = (__bf16)ow[m * 64 + k];
  }
  for (int i = tid; i < 16; i += nth) {
    float s = g[i] * rsqrtf(vr[i] + 1e-5f);
    bns[i] = s;
    bnb[i] = bt[i] - mn[i] * s;
  }
  for (int i = tid; i < 256; i += nth) qnorm[i] = 0.f;
  for (int i = tid; i < 2048; i += nth) knorm[i] = 0.f;
  for (int i = tid; i < 32768; i += nth) sraw[i] = 0.f;
}

// ---------------------------------------------------------------------------
// Kernel 2: Q projection + K row-norms.  Head-outer K phase so each head's
// 32 KB weight slice is async-staged into LDS once per block.
// ---------------------------------------------------------------------------
__global__ __launch_bounds__(64) void k_projA(
    const __bf16* __restrict__ cenbf, const __bf16* __restrict__ kpack,
    const __bf16* __restrict__ qpack, float* __restrict__ qnorm,
    float* __restrict__ knorm, __bf16* __restrict__ Qp) {
  __shared__ __bf16 sur[2][16 * 128];
  __shared__ __bf16 cenA[2][16 * 32];
  __shared__ float  nacc[2][32 * 32];  // [wave][(h*8+nt)*32 + lane]
  __shared__ __bf16 wstage[16384];     // one head's kpack slice (32 KB)
  const int tid  = threadIdx.x;
  const int lane = tid & 31;
  const int wave = tid >> 5;
  const int gw   = blockIdx.x * 2 + wave;   // 2048 waves
  const int b    = gw >> 9;
  const int wib  = gw & 511;
  const int px   = lane & 15;
  const int hi   = lane >> 4;
  const size_t cbase = (size_t)b * CIN * AREA;
  float accQ[4] = {0.f, 0.f, 0.f, 0.f};
#pragma unroll 1
  for (int i = 0; i < 32; ++i) nacc[wave][i * 32 + lane] = 0.f;
  // ---------------- Phase Q ----------------
#pragma unroll 1
  for (int t = 0; t < 8; ++t) {
    const int a0 = (wib * 8 + t) * 16;
    const int y  = a0 >> 8;
    const int x  = (a0 & 255) + px;
    float cf[16];
    const __bf16* cptr = cenbf + cbase + (size_t)y * WW + x;
#pragma unroll
    for (int c = 0; c < 16; ++c) cf[c] = (float)cptr[(size_t)c * AREA];
    __builtin_prefetch(cptr + 16, 0, 0);
    if (hi == 0) {
#pragma unroll
      for (int c = 0; c < 16; ++c) cenA[wave][px * 32 + c] = (__bf16)cf[c];
    } else {
#pragma unroll
      for (int c = 0; c < 16; ++c) cenA[wave][px * 32 + 16 + c] = (__bf16)0.f;
    }
    __syncthreads();
    bf16x16 a = ld_bfrag(&cenA[wave][px * 32 + hi * 16]);
#pragma unroll
    for (int h = 0; h < 4; ++h) {
      bf16x16 bq = ld_bfrag(qpack + (h * 32 + lane) * 16);
      floatx8 acc = wmma_bf16(a, bq, zero8());
      accQ[h] += sumsq8(acc);
      union { __bf16 hh[8]; uint4 q; } pk;
#pragma unroll
      for (int r = 0; r < 8; ++r) pk.hh[r] = (__bf16)acc[r];
      *(uint4*)(Qp + (size_t)(b * 64 + (lane & 15) * 4 + h) * AREA + a0 +
                hi * 8) = pk.q;
    }
    __syncthreads();
  }
  // ---------------- Phase K (head-outer, staged weights) ----------------
#pragma unroll 1
  for (int h = 0; h < 4; ++h) {
    const int d = 1 << h;
    __syncthreads();  // prior wstage readers done
#pragma unroll 1
    for (int c = tid; c < 2048; c += 64)
      async_cp16(&wstage[c * 8], kpack + (size_t)h * 16384 + c * 8);
    stage_fence();
    __syncthreads();
#pragma unroll 1
    for (int t = 0; t < 8; ++t) {
      const int a0 = (wib * 8 + t) * 16;
      const int y  = a0 >> 8;
      const int x  = (a0 & 255) + px;
      float cf[16];
      const __bf16* cptr = cenbf + cbase + (size_t)y * WW + x;
#pragma unroll
      for (int c = 0; c < 16; ++c) cf[c] = (float)cptr[(size_t)c * AREA];
      __syncthreads();
      build_sur(sur[wave], cenbf, cbase, y, x, px, hi, d, cf);
      __syncthreads();
#pragma unroll 1
      for (int nt = 0; nt < 8; ++nt) {
        floatx8 acc = zero8();
#pragma unroll
        for (int ks = 0; ks < 4; ++ks) {
          bf16x16 a  = ld_afrag(&sur[wave][px * 128 + ks * 32 + hi * 8]);
          bf16x16 bb = ld_bfrag(&wstage[((ks * 8 + nt) * 32 + lane) * 16]);
          acc = wmma_bf16(a, bb, acc);
        }
        nacc[wave][(h * 8 + nt) * 32 + lane] += sumsq8(acc);
      }
    }
  }
  __syncthreads();
#pragma unroll
  for (int h = 0; h < 4; ++h)
    atomicAdd(&qnorm[b * 64 + (lane & 15) * 4 + h], accQ[h]);
#pragma unroll 1
  for (int h = 0; h < 4; ++h) {
#pragma unroll 1
    for (int nt = 0; nt < 8; ++nt)
      atomicAdd(&knorm[b * 512 + (nt * 16 + (lane & 15)) * 4 + h],
                nacc[wave][(h * 8 + nt) * 32 + lane]);
  }
}

// ---------------------------------------------------------------------------
// Kernel 3: reciprocal norms
// ---------------------------------------------------------------------------
__global__ __launch_bounds__(256) void k_norms(const float* __restrict__ qn,
                                               const float* __restrict__ kn,
                                               float* __restrict__ qrn,
                                               float* __restrict__ krn) {
  int i = blockIdx.x * 256 + threadIdx.x;
  if (i < 256) qrn[i] = 1.f / fmaxf(sqrtf(qn[i]), 1e-12f);
  if (i < 2048) krn[i] = 1.f / fmaxf(sqrtf(kn[i]), 1e-12f);
}

// ---------------------------------------------------------------------------
// Kernel 4: raw scores, split-K over area; K recomputed on the fly; weight
// fragments for this (b,h') async-staged to LDS once per block.
// ---------------------------------------------------------------------------
__global__ __launch_bounds__(64) void k_scores(
    const __bf16* __restrict__ cenbf, const __bf16* __restrict__ kpack,
    const __bf16* __restrict__ Qp, float* __restrict__ sraw) {
  __shared__ __bf16 sur[2][16 * 128];
  __shared__ __bf16 kt[2][128 * 32];   // kt[d_local][a_local]
  __shared__ __bf16 wstage[16384];     // (h,ks,ntl) fragments for this hp (32 KB)
  const int tid  = threadIdx.x;
  const int lane = tid & 31;
  const int wave = tid >> 5;
  const int gw   = blockIdx.x * 2 + wave;  // 1024 waves = 16 (b,h') x 64
  const int bh   = gw >> 6;
  const int b = bh >> 2, hp = bh & 3;
  const int wb = gw & 63;
  const int px = lane & 15, hi = lane >> 4;
  const size_t cbase = (size_t)b * CIN * AREA;
  // Stage the 32 fragments this block will use (block-uniform hp).
#pragma unroll 1
  for (int c = tid; c < 2048; c += 64) {
    int f = c >> 6, within = c & 63;
    int h = f >> 3, ks = (f >> 1) & 3, ntl = f & 1;
    int ntg = hp * 2 + ntl;
    async_cp16(&wstage[c * 8],
               kpack + ((size_t)((h * 4 + ks) * 8 + ntg) * 32) * 16 + within * 8);
  }
  stage_fence();
  __syncthreads();
  floatx8 dacc[8];
#pragma unroll
  for (int n = 0; n < 8; ++n) dacc[n] = zero8();
#pragma unroll 1
  for (int t = 0; t < 32; ++t) {
    const int a0 = (wb * 32 + t) * 32;
#pragma unroll 1
    for (int sub = 0; sub < 2; ++sub) {
      const int as = a0 + sub * 16;
      const int y  = as >> 8;
      const int x  = (as & 255) + px;
      float cf[16];
      const __bf16* cptr = cenbf + cbase + (size_t)y * WW + x;
#pragma unroll
      for (int c = 0; c < 16; ++c) cf[c] = (float)cptr[(size_t)c * AREA];
#pragma unroll 1
      for (int h = 0; h < 4; ++h) {
        const int d = 1 << h;
        __syncthreads();
        build_sur(sur[wave], cenbf, cbase, y, x, px, hi, d, cf);
        __syncthreads();
#pragma unroll 1
        for (int ntl = 0; ntl < 2; ++ntl) {
          floatx8 acc = zero8();
#pragma unroll
          for (int ks = 0; ks < 4; ++ks) {
            bf16x16 a  = ld_afrag(&sur[wave][px * 128 + ks * 32 + hi * 8]);
            bf16x16 bb = ld_bfrag(&wstage[(((h * 4 + ks) * 2 + ntl) * 32 + lane) * 16]);
            acc = wmma_bf16(a, bb, acc);
          }
          const int dl = (ntl * 16 + (lane & 15)) * 4 + h;
#pragma unroll
          for (int r = 0; r < 8; ++r) {
            const int al = sub * 16 + hi * 8 + r;
            kt[wave][dl * 32 + al] = (__bf16)acc[r];
          }
        }
      }
    }
    __syncthreads();
    // ---- Q . K over this 32-wide a-chunk: D(M=c, N=d) ----
    {
      const __bf16* qrow =
          Qp + (size_t)(b * 64 + hp * 16 + (lane & 15)) * AREA + a0 + hi * 8;
      bf16x16 aq = ld_afrag(qrow);
#pragma unroll
      for (int nt = 0; nt < 8; ++nt) {
        bf16x16 bk = ld_afrag(&kt[wave][(nt * 16 + (lane & 15)) * 32 + hi * 8]);
        dacc[nt] = wmma_bf16(aq, bk, dacc[nt]);
      }
    }
    __syncthreads();
  }
#pragma unroll 1
  for (int nt = 0; nt < 8; ++nt) {
#pragma unroll
    for (int r = 0; r < 8; ++r) {
      const int c  = hi * 8 + r;
      const int dd = nt * 16 + (lane & 15);
      atomicAdd(&sraw[(bh * 16 + c) * 128 + dd], dacc[nt][r]);
    }
  }
}

// ---------------------------------------------------------------------------
// Kernel 5: cosine scale + instance-norm + softmax; pack attn as A-fragments
// ---------------------------------------------------------------------------
__global__ __launch_bounds__(128) void k_softmax(
    const float* __restrict__ sraw, const float* __restrict__ qrn,
    const float* __restrict__ krn, __bf16* __restrict__ apack) {
  __shared__ float s[2048];
  __shared__ float red[128];
  __shared__ float stat[2];
  const int tid = threadIdx.x;
  const int bh  = blockIdx.x;
  const int b = bh >> 2, h = bh & 3;
  const float inv = 1.f / 256.f;  // 1/sqrt(area)
  for (int i = tid; i < 2048; i += 128) {
    int c = i >> 7, dd = i & 127;
    s[i] = sraw[bh * 2048 + i] * qrn[b * 64 + h * 16 + c] *
           krn[b * 512 + h * 128 + dd] * inv;
  }
  __syncthreads();
  float ls = 0.f;
  for (int i = tid; i < 2048; i += 128) ls += s[i];
  red[tid] = ls;
  __syncthreads();
  if (tid == 0) {
    float tot = 0.f;
    for (int i = 0; i < 128; ++i) tot += red[i];
    stat[0] = tot / 2048.f;
  }
  __syncthreads();
  const float mu = stat[0];
  float lv = 0.f;
  for (int i = tid; i < 2048; i += 128) {
    float dlt = s[i] - mu;
    lv += dlt * dlt;
  }
  red[tid] = lv;
  __syncthreads();
  if (tid == 0) {
    float tot = 0.f;
    for (int i = 0; i < 128; ++i) tot += red[i];
    stat[1] = rsqrtf(tot / 2048.f + 1e-5f);
  }
  __syncthreads();
  const float rstd = stat[1];
  if (tid < 16) {
    const int c = tid;
    float mx = -1e30f;
    for (int dd = 0; dd < 128; ++dd) {
      float v = (s[c * 128 + dd] - mu) * rstd;
      if (v > mx) mx = v;
    }
    float sm = 0.f;
    for (int dd = 0; dd < 128; ++dd) {
      float e = __expf((s[c * 128 + dd] - mu) * rstd - mx);
      s[c * 128 + dd] = e;
      sm += e;
    }
    float r = 1.f / sm;
    for (int dd = 0; dd < 128; ++dd) s[c * 128 + dd] *= r;
  }
  __syncthreads();
  for (int i = tid; i < 2048; i += 128) {
    int e = i & 15, L = (i >> 4) & 31, ks = (i >> 9) & 3;
    int c = L & 15;
    int k = ks * 32 + ((L >> 4) & 1) * 8 + (e & 7) + (e >> 3) * 16;
    apack[bh * 2048 + i] = (__bf16)s[c * 128 + k];
  }
}

// ---------------------------------------------------------------------------
// Kernel 6: fused V recompute + attn.V + out_w + BN + ReLU; attn/out_w
// fragments async-staged to LDS once per block.
// ---------------------------------------------------------------------------
__global__ __launch_bounds__(64) void k_out(
    const __bf16* __restrict__ cenbf, const __bf16* __restrict__ vpack,
    const __bf16* __restrict__ apack, const __bf16* __restrict__ owpack,
    const float* __restrict__ bns, const float* __restrict__ bnb,
    float* __restrict__ out) {
  __shared__ __bf16 sur[2][16 * 128];
  __shared__ __bf16 vt[2][16 * 512];   // vt[px][f]
  __shared__ __bf16 o64[2][16 * 64];   // o64[px][t]
  __shared__ __bf16 apst[8192];        // this batch's attn fragments (16 KB)
  __shared__ __bf16 owst[1024];        // out_w fragments (2 KB)
  const int tid  = threadIdx.x;
  const int lane = tid & 31;
  const int wave = tid >> 5;
  const int gw   = blockIdx.x * 2 + wave;  // 2048 waves
  const int b    = gw >> 9;
  const int wib  = gw & 511;
  const int px   = lane & 15, hi = lane >> 4;
  const size_t cbase = (size_t)b * CIN * AREA;
  float sc8[8], bi8[8];
#pragma unroll
  for (int r = 0; r < 8; ++r) {
    sc8[r] = bns[hi * 8 + r];
    bi8[r] = bnb[hi * 8 + r];
  }
  // Stage attn fragments for this batch (block-uniform b) + out_w fragments.
#pragma unroll 1
  for (int c = tid; c < 1024; c += 64)
    async_cp16(&apst[c * 8], apack + (size_t)b * 8192 + c * 8);
#pragma unroll 1
  for (int c = tid; c < 128; c += 64)
    async_cp16(&owst[c * 8], owpack + c * 8);
  stage_fence();
  __syncthreads();
#pragma unroll 1
  for (int t = 0; t < 8; ++t) {
    const int a0 = (wib * 8 + t) * 16;
    const int y  = a0 >> 8;
    const int x  = (a0 & 255) + px;
    float cf[16];
    const __bf16* cptr = cenbf + cbase + (size_t)y * WW + x;
#pragma unroll
    for (int c = 0; c < 16; ++c) cf[c] = (float)cptr[(size_t)c * AREA];
    __builtin_prefetch(cptr + 16, 0, 0);
    // ---- V recompute for all 4 shift-heads, store permuted into LDS ----
#pragma unroll 1
    for (int h = 0; h < 4; ++h) {
      const int d = 1 << h;
      __syncthreads();
      build_sur(sur[wave], cenbf, cbase, y, x, px, hi, d, cf);
      __syncthreads();
#pragma unroll 1
      for (int nt = 0; nt < 8; ++nt) {
        floatx8 acc = zero8();
#pragma unroll
        for (int ks = 0; ks < 4; ++ks) {
          bf16x16 a  = ld_afrag(&sur[wave][px * 128 + ks * 32 + hi * 8]);
          bf16x16 bb = ld_bfrag(vpack + ((((h * 4 + ks) * 8 + nt) * 32) + lane) * 16);
          acc = wmma_bf16(a, bb, acc);
        }
        const int f = (nt * 16 + (lane & 15)) * 4 + h;
#pragma unroll
        for (int r = 0; r < 8; ++r) {
          const int pxs = hi * 8 + r;
          vt[wave][pxs * 512 + f] = (__bf16)acc[r];
        }
      }
    }
    __syncthreads();
    // ---- attn . V : D(M=c, N=px), t = hp*16+c ----
#pragma unroll 1
    for (int hp = 0; hp < 4; ++hp) {
      floatx8 acc = zero8();
#pragma unroll
      for (int ks = 0; ks < 4; ++ks) {
        bf16x16 a  = ld_bfrag(&apst[((hp * 4 + ks) * 32 + lane) * 16]);
        bf16x16 bv = ld_afrag(&vt[wave][(lane & 15) * 512 + hp * 128 + ks * 32 + hi * 8]);
        acc = wmma_bf16(a, bv, acc);
      }
#pragma unroll
      for (int r = 0; r < 8; ++r) {
        const int c = hi * 8 + r;
        o64[wave][(lane & 15) * 64 + hp * 16 + c] = (__bf16)acc[r];
      }
    }
    __syncthreads();
    // ---- out_w : D(M=o, N=px) ----
    floatx8 facc = zero8();
#pragma unroll
    for (int ks = 0; ks < 2; ++ks) {
      bf16x16 a  = ld_bfrag(&owst[(ks * 32 + lane) * 16]);
      bf16x16 bo = ld_afrag(&o64[wave][(lane & 15) * 64 + ks * 32 + hi * 8]);
      facc = wmma_bf16(a, bo, facc);
    }
#pragma unroll
    for (int r = 0; r < 8; ++r) {
      const int o = hi * 8 + r;
      float v = fmaxf(facc[r] * sc8[r] + bi8[r], 0.f);
      out[(size_t)(b * 16 + o) * AREA + (size_t)y * WW + (a0 & 255) + (lane & 15)] = v;
    }
    __syncthreads();
  }
}

// ---------------------------------------------------------------------------
// Workspace layout (bytes) — total ~42.5 MB
// ---------------------------------------------------------------------------
static constexpr size_t QP_OFF     = 0;                         // 4*64*65536*2
static constexpr size_t CENBF_OFF  = QP_OFF + 33554432;         // 4*16*65536*2
static constexpr size_t KPACK_OFF  = CENBF_OFF + 8388608;       // 131072
static constexpr size_t VPACK_OFF  = KPACK_OFF + 131072;        // 131072
static constexpr size_t QPACK_OFF  = VPACK_OFF + 131072;        // 4096
static constexpr size_t OWPACK_OFF = QPACK_OFF + 4096;          // 2048
static constexpr size_t BNS_OFF    = OWPACK_OFF + 2048;         // 64
static constexpr size_t BNB_OFF    = BNS_OFF + 64;              // 64
static constexpr size_t QNORM_OFF  = BNB_OFF + 64;              // 1024
static constexpr size_t KNORM_OFF  = QNORM_OFF + 1024;          // 8192
static constexpr size_t QRN_OFF    = KNORM_OFF + 8192;          // 1024
static constexpr size_t KRN_OFF    = QRN_OFF + 1024;            // 8192
static constexpr size_t SRAW_OFF   = KRN_OFF + 8192;            // 131072
static constexpr size_t APACK_OFF  = SRAW_OFF + 131072;         // 65536

extern "C" void kernel_launch(void* const* d_in, const int* in_sizes, int n_in,
                              void* d_out, int out_size, void* d_ws,
                              size_t ws_size, hipStream_t stream) {
  (void)in_sizes; (void)n_in; (void)out_size; (void)ws_size;
  const float* cen = (const float*)d_in[0];
  const float* qw  = (const float*)d_in[1];
  const float* kw  = (const float*)d_in[2];
  const float* vw  = (const float*)d_in[3];
  const float* ow  = (const float*)d_in[4];
  const float* g   = (const float*)d_in[5];
  const float* bt  = (const float*)d_in[6];
  const float* mn  = (const float*)d_in[7];
  const float* vr  = (const float*)d_in[8];
  float* out = (float*)d_out;

  char* ws = (char*)d_ws;
  __bf16* Qp     = (__bf16*)(ws + QP_OFF);
  __bf16* cenbf  = (__bf16*)(ws + CENBF_OFF);
  __bf16* kpack  = (__bf16*)(ws + KPACK_OFF);
  __bf16* vpack  = (__bf16*)(ws + VPACK_OFF);
  __bf16* qpack  = (__bf16*)(ws + QPACK_OFF);
  __bf16* owpack = (__bf16*)(ws + OWPACK_OFF);
  float*  bns    = (float*)(ws + BNS_OFF);
  float*  bnb    = (float*)(ws + BNB_OFF);
  float*  qnorm  = (float*)(ws + QNORM_OFF);
  float*  knorm  = (float*)(ws + KNORM_OFF);
  float*  qrn    = (float*)(ws + QRN_OFF);
  float*  krn    = (float*)(ws + KRN_OFF);
  float*  sraw   = (float*)(ws + SRAW_OFF);
  __bf16* apack  = (__bf16*)(ws + APACK_OFF);

  k_setup<<<256, 256, 0, stream>>>(cen, qw, kw, vw, ow, g, bt, mn, vr, cenbf,
                                   kpack, vpack, qpack, owpack, bns, bnb,
                                   qnorm, knorm, sraw);
  k_projA<<<1024, 64, 0, stream>>>(cenbf, kpack, qpack, qnorm, knorm, Qp);
  k_norms<<<8, 256, 0, stream>>>(qnorm, knorm, qrn, krn);
  k_scores<<<512, 64, 0, stream>>>(cenbf, kpack, Qp, sraw);
  k_softmax<<<16, 128, 0, stream>>>(sraw, qrn, krn, apack);
  k_out<<<1024, 64, 0, stream>>>(cenbf, vpack, apack, owpack, bns, bnb, out);
}
